// GRNNModel_47691316855467
// MI455X (gfx1250) — matrix operand order repaired and checked by reading, not compiled, not run on hardware
//
#include <hip/hip_runtime.h>
#include <stdint.h>

// ---- problem constants (fixed by reference) ----
#define BN   64      // batch
#define TT   12      // encoder steps
#define FF   2       // features
#define NN   2048    // nodes
#define HH   64      // hidden
#define PP   6       // horizon
#define NBC  (BN*HH) // 4096 columns of diffusion GEMM
#define NROW (NN*BN) // 131072 rows of [node,batch] state

typedef __attribute__((ext_vector_type(16))) __bf16 v16bf;
typedef __attribute__((ext_vector_type(8)))  float  v8f;

union BFrag {
    v16bf v;
    unsigned int u[8];
    unsigned short s[16];
};

__device__ __forceinline__ unsigned short f2bf(float f) {
    union { float f; uint32_t u; } a; a.f = f;
    uint32_t u = a.u;
    return (unsigned short)((u + 0x7FFFu + ((u >> 16) & 1u)) >> 16); // RNE truncation
}

__device__ __forceinline__ float sigm(float x) {
    return 1.0f / (1.0f + __expf(-x));
}

// ------------------------------------------------------------------
// init: AT_bf16[m][n] = bf16(adj[n][m])   (2048x2048)
// ------------------------------------------------------------------
__global__ __launch_bounds__(256) void k_adjT(const float* __restrict__ adj,
                                              unsigned short* __restrict__ AT) {
    int idx = blockIdx.x * 256 + threadIdx.x;
    if (idx >= NN * NN) return;
    int m = idx >> 11, n = idx & (NN - 1);
    AT[idx] = f2bf(adj[(size_t)n * NN + m]);
}

// ------------------------------------------------------------------
// init state: X[n*B+b][h] = hidden0[b][h][n], f32 + bf16 copies
// ------------------------------------------------------------------
__global__ __launch_bounds__(256) void k_initH(const float* __restrict__ h0,
                                               float* __restrict__ Xf,
                                               unsigned short* __restrict__ Xb) {
    int idx = blockIdx.x * 256 + threadIdx.x;
    if (idx >= NROW * HH) return;
    int h   = idx & 63;
    int row = idx >> 6;
    int nn  = row >> 6;
    int b   = row & 63;
    float v = h0[(size_t)b * HH * NN + (size_t)h * NN + nn];
    Xf[idx] = v;
    Xb[idx] = f2bf(v);
}

// ------------------------------------------------------------------
// diffusion GEMM: HT[2048,4096] = AT[2048,2048](bf16) @ Xb[2048,4096](bf16)
// block = 256 thr (8 waves); tile M=128 (16 rows/wave), tile N=64 (4 ct/wave)
// ------------------------------------------------------------------
__global__ __launch_bounds__(256) void k_diffuse(const unsigned short* __restrict__ AT,
                                                 const unsigned short* __restrict__ Xb,
                                                 float* __restrict__ HT) {
    __shared__ unsigned short Atile[128 * 32]; // [m][k]
    __shared__ unsigned short Xtile[64 * 32];  // [n][k] (transposed: K contiguous)

    const int tid  = threadIdx.x;
    const int lane = tid & 31;
    const int wave = tid >> 5;
    const int bm   = blockIdx.y;  // 0..15
    const int bn   = blockIdx.x;  // 0..63
    const int mrow = lane & 15;
    const bool lo  = lane < 16;

    v8f acc[4];
#pragma unroll
    for (int i = 0; i < 4; i++) acc[i] = v8f{0.f,0.f,0.f,0.f,0.f,0.f,0.f,0.f};

    for (int k0 = 0; k0 < NN; k0 += 32) {
        // stage A tile (128x32 bf16) as 2048 dword moves
#pragma unroll
        for (int i = 0; i < 8; i++) {
            int idx = i * 256 + tid;            // 0..2047
            int r   = idx >> 4;                 // row 0..127
            int kp  = (idx & 15) << 1;          // k = 0,2,..,30
            const unsigned short* g = AT + (size_t)(bm * 128 + r) * NN + k0 + kp;
            *(unsigned int*)(Atile + r * 32 + kp) = *(const unsigned int*)g;
            __builtin_prefetch(g + 32, 0, 3);   // next K-tile of A stream
        }
        // stage X tile transposed: global [k][n] -> LDS [n][k]
#pragma unroll
        for (int i = 0; i < 4; i++) {
            int idx = i * 256 + tid;            // 0..1023
            int k   = idx >> 5;                 // 0..31
            int np  = (idx & 31) << 1;          // n = 0,2,..,62
            unsigned int v = *(const unsigned int*)(Xb + (size_t)(k0 + k) * NBC + bn * 64 + np);
            Xtile[np * 32 + k]       = (unsigned short)(v & 0xFFFFu);
            Xtile[(np + 1) * 32 + k] = (unsigned short)(v >> 16);
        }
        __syncthreads();

        // A fragment (16x32 bf16, ISA per-lane layout)
        BFrag af;
#pragma unroll
        for (int j = 0; j < 8; j++) {
            int ks = (j < 4) ? (2*j + (lo ? 0 : 8)) : (2*(j-4) + (lo ? 16 : 24));
            af.u[j] = *(const unsigned int*)(Atile + (wave * 16 + mrow) * 32 + ks);
        }
#pragma unroll
        for (int ct = 0; ct < 4; ct++) {
            BFrag bf;
#pragma unroll
            for (int j = 0; j < 8; j++) {
                int k = (j < 4) ? (2*j + (lo ? 0 : 8)) : (2*(j-4) + (lo ? 16 : 24));
                bf.u[j] = *(const unsigned int*)(Xtile + (ct * 16 + mrow) * 32 + k);
            }
            acc[ct] = __builtin_amdgcn_wmma_f32_16x16x32_bf16(
                false, af.v, false, bf.v, (short)0, acc[ct], false, false);
        }
        __syncthreads();
    }

    // C layout: VGPR v -> row v (lanes 0-15) / v+8 (lanes 16-31); col = lane%16 + ct*16
    const int rbase = bm * 128 + wave * 16 + (lo ? 0 : 8);
#pragma unroll
    for (int ct = 0; ct < 4; ct++) {
#pragma unroll
        for (int v = 0; v < 8; v++) {
            HT[(size_t)(rbase + v) * NBC + bn * 64 + ct * 16 + mrow] = acc[ct][v];
        }
    }
}

// ------------------------------------------------------------------
// fused gates: 3x (HTrows[NROW,64] @ U^T[64,64]) via WMMA + GRU epilogue
// block = 256 thr (8 waves); each wave owns a 16-row tile -> 128 rows/block
// writes new state Xf (f32) and Xb (bf16)
// ------------------------------------------------------------------
__global__ __launch_bounds__(256) void k_gates(
    const float* __restrict__ HT,
    float* __restrict__ Xf, unsigned short* __restrict__ Xb,
    const float* __restrict__ Urw, const float* __restrict__ Uzw, const float* __restrict__ Unw,
    const float* __restrict__ Urb, const float* __restrict__ Uzb, const float* __restrict__ Unb,
    const float* __restrict__ Wrw, const float* __restrict__ Wzw, const float* __restrict__ Wnw,
    const float* __restrict__ Wrb, const float* __restrict__ Wzb, const float* __restrict__ Wnb,
    const float* __restrict__ x, const float* __restrict__ dec, int t, int mode) {
    __shared__ unsigned short Ub[3 * 64 * 64]; // bf16 of U_g row-major = B[k-major per row n]

    const int tid = threadIdx.x;
    for (int i = tid; i < 3 * 4096; i += 256) {
        int g = i >> 12, e = i & 4095;
        const float* U = (g == 0) ? Urw : (g == 1) ? Uzw : Unw;
        Ub[i] = f2bf(U[e]);
    }
    __syncthreads();

    const int lane  = tid & 31;
    const int wave  = tid >> 5;
    const int mrow  = lane & 15;
    const bool lo   = lane < 16;
    const int rbase = blockIdx.x * 128 + wave * 16;

    v8f acc[3][4];
#pragma unroll
    for (int g = 0; g < 3; g++)
#pragma unroll
        for (int c = 0; c < 4; c++) acc[g][c] = v8f{0.f,0.f,0.f,0.f,0.f,0.f,0.f,0.f};

#pragma unroll
    for (int kc = 0; kc < 2; kc++) {
        BFrag af;
#pragma unroll
        for (int j = 0; j < 8; j++) {
            int ks = kc * 32 + ((j < 4) ? (2*j + (lo ? 0 : 8)) : (2*(j-4) + (lo ? 16 : 24)));
            const float* p = HT + (size_t)(rbase + mrow) * HH + ks;
            af.s[2*j]   = f2bf(p[0]);
            af.s[2*j+1] = f2bf(p[1]);
        }
#pragma unroll
        for (int g = 0; g < 3; g++) {
#pragma unroll
            for (int ct = 0; ct < 4; ct++) {
                BFrag bf;
#pragma unroll
                for (int j = 0; j < 8; j++) {
                    int k = kc * 32 + ((j < 4) ? (2*j + (lo ? 0 : 8)) : (2*(j-4) + (lo ? 16 : 24)));
                    bf.u[j] = *(const unsigned int*)(Ub + g * 4096 + (ct * 16 + mrow) * 64 + k);
                }
                acc[g][ct] = __builtin_amdgcn_wmma_f32_16x16x32_bf16(
                    false, af.v, false, bf.v, (short)0, acc[g][ct], false, false);
            }
        }
    }

    // GRU epilogue
#pragma unroll
    for (int ct = 0; ct < 4; ct++) {
        const int hcol = ct * 16 + mrow;
        const float urb = Urb[hcol], uzb = Uzb[hcol], unb = Unb[hcol];
        const float wr0 = Wrw[hcol*2], wr1 = Wrw[hcol*2+1], wrb = Wrb[hcol];
        const float wz0 = Wzw[hcol*2], wz1 = Wzw[hcol*2+1], wzb = Wzb[hcol];
        const float wn0 = Wnw[hcol*2], wn1 = Wnw[hcol*2+1], wnb = Wnb[hcol];
#pragma unroll
        for (int v = 0; v < 8; v++) {
            int row = rbase + v + (lo ? 0 : 8);
            int nn  = row >> 6;
            int b   = row & 63;
            float i0, i1;
            if (mode == 0) { // encoder: x[b, t, f*N + n]
                const float* xp = x + (size_t)b * (TT * FF * NN) + (size_t)t * (FF * NN) + nn;
                i0 = xp[0];
                i1 = xp[NN];
            } else {         // decoder: inp = (out, 0)
                i0 = dec[(size_t)b * NN + nn];
                i1 = 0.f;
            }
            float rv  = sigm(acc[0][ct][v] + urb + wr0 * i0 + wr1 * i1 + wrb);
            float zv  = sigm(acc[1][ct][v] + uzb + wz0 * i0 + wz1 * i1 + wzb);
            float nv  = tanhf(rv * (acc[2][ct][v] + unb) + wn0 * i0 + wn1 * i1 + wnb);
            float htv = HT[(size_t)row * HH + hcol];
            float hnew = (1.f - zv) * nv + zv * htv;
            Xf[(size_t)row * HH + hcol] = hnew;
            Xb[(size_t)row * HH + hcol] = f2bf(hnew);
        }
    }
}

// ------------------------------------------------------------------
// decoder readout: out[b,p,n] = dot(X[n*B+b], fc_w) + fc_b ; also feeds dec buf
// ------------------------------------------------------------------
__global__ __launch_bounds__(256) void k_decout(const float* __restrict__ Xf,
                                                const float* __restrict__ fcw,
                                                const float* __restrict__ fcb,
                                                float* __restrict__ out,
                                                float* __restrict__ dec, int p) {
    int idx = blockIdx.x * 256 + threadIdx.x;
    if (idx >= NROW) return;
    int nn = idx >> 6, b = idx & 63;
    float a = fcb[0];
    const float* r = Xf + (size_t)idx * HH;
#pragma unroll
    for (int h = 0; h < HH; h++) a += r[h] * fcw[h];
    out[(size_t)b * (PP * NN) + (size_t)p * NN + nn] = a;
    dec[(size_t)b * NN + nn] = a;
}

// ------------------------------------------------------------------
extern "C" void kernel_launch(void* const* d_in, const int* in_sizes, int n_in,
                              void* d_out, int out_size, void* d_ws, size_t ws_size,
                              hipStream_t stream) {
    const float* x    = (const float*)d_in[0];
    const float* h0   = (const float*)d_in[1];
    const float* adj  = (const float*)d_in[2];
    const float* Urw  = (const float*)d_in[3];
    const float* Urb  = (const float*)d_in[4];
    const float* Wrw  = (const float*)d_in[5];
    const float* Wrb  = (const float*)d_in[6];
    const float* Uzw  = (const float*)d_in[7];
    const float* Uzb  = (const float*)d_in[8];
    const float* Wzw  = (const float*)d_in[9];
    const float* Wzb  = (const float*)d_in[10];
    const float* Unw  = (const float*)d_in[11];
    const float* Unb  = (const float*)d_in[12];
    const float* Wnw  = (const float*)d_in[13];
    const float* Wnb  = (const float*)d_in[14];
    const float* fcw  = (const float*)d_in[15];
    const float* fcb  = (const float*)d_in[16];
    float* out = (float*)d_out;

    char* ws = (char*)d_ws;
    size_t off = 0;
    unsigned short* AT = (unsigned short*)(ws + off); off += (size_t)NN * NN * 2;        // 8.4 MB
    float*          Xf = (float*)(ws + off);          off += (size_t)NROW * HH * 4;      // 33.6 MB
    unsigned short* Xb = (unsigned short*)(ws + off); off += (size_t)NROW * HH * 2;      // 16.8 MB
    float*          HT = (float*)(ws + off);          off += (size_t)NROW * HH * 4;      // 33.6 MB
    float*          dec = (float*)(ws + off);         off += (size_t)NN * BN * 4;        // 0.5 MB

    // ---- init ----
    k_adjT<<<(NN * NN + 255) / 256, 256, 0, stream>>>(adj, AT);
    k_initH<<<(NROW * HH + 255) / 256, 256, 0, stream>>>(h0, Xf, Xb);

    dim3 gDiff(NBC / 64, NN / 128);   // 64 x 16
    int  gGates = NROW / 128;         // 1024

    // ---- encoder: 12 GRU steps ----
    for (int t = 0; t < TT; t++) {
        k_diffuse<<<gDiff, 256, 0, stream>>>(AT, Xb, HT);
        k_gates<<<gGates, 256, 0, stream>>>(HT, Xf, Xb,
                                            Urw, Uzw, Unw, Urb, Uzb, Unb,
                                            Wrw, Wzw, Wnw, Wrb, Wzb, Wnb,
                                            x, nullptr, t, 0);
    }
    // ---- decoder: 6 steps of (readout, GRU) ----
    for (int p = 0; p < PP; p++) {
        k_decout<<<(NROW + 255) / 256, 256, 0, stream>>>(Xf, fcw, fcb, out, dec, p);
        k_diffuse<<<gDiff, 256, 0, stream>>>(AT, Xb, HT);
        k_gates<<<gGates, 256, 0, stream>>>(HT, Xf, Xb,
                                            Urw, Uzw, Unw, Urb, Uzb, Unb,
                                            Wrw, Wzw, Wnw, Wrb, Wzb, Wnb,
                                            nullptr, dec, 0, 1);
    }
}
